// Decoder_27960237097243
// MI455X (gfx1250) — matrix operand order repaired and checked by reading, not compiled, not run on hardware
//
#include <hip/hip_runtime.h>
#include <math.h>

// MI455X / gfx1250 — wave32, WMMA. One workgroup (8 waves) per batch element.
// Weights fp32->bf16 once into VGPRs as WMMA B-operands for all T steps; only
// the 256-element activation vector round-trips LDS. 4 barriers per step.
// Prop layers use an explicit 2-deep software pipeline (3 rotating A quads)
// so ds_load_b128 issue runs 2 iterations ahead of the consuming WMMA pair.

typedef __attribute__((ext_vector_type(16))) __bf16 v16bf;
typedef __attribute__((ext_vector_type(8)))  __bf16 v8bf;
typedef __attribute__((ext_vector_type(8)))  float  v8f;

#define C_DIM 16
#define H_DIM 256

// Build A operand (16x32 bf16, all 16 rows replicated with h[k0..k0+31]).
// A-layout (matches B): lane<16 needs K=[k0..k0+15],
// lane>=16 needs K=[k0+16..k0+31]  -> s = (lane>>4)*8, offset 2*s.
__device__ __forceinline__ v16bf load_a(const __bf16* h, int k0, int s) {
    v8bf lo = *(const v8bf*)(h + k0 + 2 * s);
    v8bf hi = *(const v8bf*)(h + k0 + 2 * s + 8);
    return __builtin_shufflevector(lo, hi, 0,1,2,3,4,5,6,7,8,9,10,11,12,13,14,15);
}

__device__ __forceinline__ v8f wmma_bf16(v16bf a, v16bf b, v8f c) {
    return __builtin_amdgcn_wmma_f32_16x16x32_bf16(false, a, false, b,
                                                   (short)0, c, false, false);
}

__device__ __forceinline__ v8f splat8(float x) {
    v8f r = {x, x, x, x, x, x, x, x};
    return r;
}

// One 256->32 slice of a hidden layer: dual interleaved WMMA K-chains over the
// wave's two owned N-tiles, with A-operand loads pipelined 2 deep.
__device__ __forceinline__ void hidden_layer(const __bf16* __restrict__ src,
                                             __bf16* __restrict__ dst,
                                             const v16bf Bw[2][8],
                                             float bias0, float bias1,
                                             int wave, int col, int s) {
    v8f a0 = splat8(bias0);
    v8f a1 = splat8(bias1);
    v16bf a_cur = load_a(src, 0,  s);
    v16bf a_nxt = load_a(src, 32, s);
    #pragma unroll
    for (int kt = 0; kt < 8; ++kt) {
        v16bf a_pre;
        if (kt < 6) a_pre = load_a(src, 32 * (kt + 2), s);
        a0 = wmma_bf16(a_cur, Bw[0][kt], a0);
        a1 = wmma_bf16(a_cur, Bw[1][kt], a1);
        a_cur = a_nxt;
        a_nxt = a_pre;
    }
    float v0 = a0[0]; v0 = v0 > 0.0f ? v0 : 0.0f;
    float v1 = a1[0]; v1 = v1 > 0.0f ? v1 : 0.0f;
    // rows replicated: lanes L and L+16 store identical values
    dst[32 * wave + col]      = (__bf16)v0;
    dst[32 * wave + 16 + col] = (__bf16)v1;
}

__global__ __launch_bounds__(256, 1)
void decoder_rollout_kernel(const float* __restrict__ y0,
                            const float* __restrict__ inW,
                            const float* __restrict__ inB,
                            const float* __restrict__ outW,
                            const float* __restrict__ outB,
                            const float* __restrict__ pW,
                            const float* __restrict__ pB,
                            const float* __restrict__ cutoffp,
                            const int*   __restrict__ Tp,
                            float* __restrict__ out) {
    const int b     = blockIdx.x;
    const int tid   = threadIdx.x;
    const int lane  = tid & 31;
    const int wave  = tid >> 5;         // 0..7
    const int col   = lane & 15;        // B/C/D column within tile
    const int khalf = (lane >> 4) * 16; // B-operand K-half select
    const int s     = (lane >> 4) * 8;  // A-operand K-half select
    const int   T   = Tp[0];
    const float cut = cutoffp[0];
    const float dt_over_cut = 1e-6f / cut;

    __shared__ __align__(16) __bf16 hA[H_DIM];
    __shared__ __align__(16) __bf16 hB[H_DIM];
    __shared__ __align__(16) __bf16 ypad[32];           // y bf16, K padded to 32
    __shared__ __align__(16) float  out_part[C_DIM][8]; // [c][wave] split-K partials

    const float* inWb  = inW  + (size_t)b * C_DIM * H_DIM;
    const float* pW0   = pW   + (size_t)b * 2 * H_DIM * H_DIM;
    const float* pW1   = pW0  + H_DIM * H_DIM;
    const float* outWb = outW + (size_t)b * H_DIM * C_DIM;

    // ---- One-time: stage all weights into VGPRs as bf16 WMMA B-operands ----
    // Wave w owns output columns [32w, 32w+32) for in/prop layers (2 N-tiles),
    // and K-chunk [32w, 32w+32) of the split-K output layer.
    v16bf Bin[2], Bp0[2][8], Bp1[2][8], Bout;
    float biasIn[2], biasP0[2], biasP1[2];

    #pragma unroll
    for (int i = 0; i < 2; ++i) {
        const int n = 32 * wave + 16 * i + col;
        biasIn[i] = inB[(size_t)b * H_DIM + n];
        biasP0[i] = pB[(size_t)b * 2 * H_DIM + n];
        biasP1[i] = pB[(size_t)b * 2 * H_DIM + H_DIM + n];
        #pragma unroll
        for (int e = 0; e < 16; ++e) {   // B element e <-> K = khalf + e
            const int K = khalf + e;
            Bin[i][e] = (K < C_DIM) ? (__bf16)inWb[K * H_DIM + n] : (__bf16)0.0f;
        }
        #pragma unroll
        for (int kt = 0; kt < 8; ++kt) {
            #pragma unroll
            for (int e = 0; e < 16; ++e) {
                const int K = 32 * kt + khalf + e;
                Bp0[i][kt][e] = (__bf16)pW0[K * H_DIM + n];
                Bp1[i][kt][e] = (__bf16)pW1[K * H_DIM + n];
            }
        }
    }
    #pragma unroll
    for (int e = 0; e < 16; ++e) {
        const int K = 32 * wave + khalf + e;
        Bout[e] = (__bf16)outWb[K * C_DIM + col];
    }
    const float ob = outB[(size_t)b * C_DIM + col];

    // y state lives in a register per lane (value for channel c = col),
    // maintained redundantly & identically by every wave.
    float yreg = y0[(size_t)b * C_DIM + col];

    // Every wave writes the whole ypad itself (identical values across waves
    // -> benign overlap); its own later reads are ordered by DScnt.
    if (lane < 16) ypad[lane] = (__bf16)yreg;
    else           ypad[lane] = (__bf16)0.0f;   // K padding (16 -> 32), constant
    __syncthreads();

    // ---------------------- sequential time rollout ----------------------
    for (int t = 0; t < T; ++t) {
        // in-layer: hA = relu(y * Win + bIn)   (K=32 padded, 1 WMMA per tile)
        {
            v16bf a = load_a(ypad, 0, s);
            v8f a0 = wmma_bf16(a, Bin[0], splat8(biasIn[0]));
            v8f a1 = wmma_bf16(a, Bin[1], splat8(biasIn[1]));
            float v0 = a0[0]; v0 = v0 > 0.0f ? v0 : 0.0f;
            float v1 = a1[0]; v1 = v1 > 0.0f ? v1 : 0.0f;
            hA[32 * wave + col]      = (__bf16)v0;
            hA[32 * wave + 16 + col] = (__bf16)v1;
        }
        __syncthreads();

        // prop layer 0: hB = relu(hA * W0 + b0)
        hidden_layer(hA, hB, Bp0, biasP0[0], biasP0[1], wave, col, s);
        __syncthreads();

        // prop layer 1: hA = relu(hB * W1 + b1)
        hidden_layer(hB, hA, Bp1, biasP1[0], biasP1[1], wave, col, s);
        __syncthreads();

        // out-layer: split-K over 8 waves; wave w writes partial for its K-chunk
        {
            v8f acc = wmma_bf16(load_a(hA, 32 * wave, s), Bout, splat8(0.0f));
            out_part[col][wave] = acc[0];
        }
        __syncthreads();

        // y update: ALL waves reduce the 8 partials and update y redundantly
        // (identical results). No trailing barrier needed: each wave's in-layer
        // read of ypad is covered by its own write (same values across waves).
        {
            v8f p = *(const v8f*)&out_part[col][0];
            float o = ((p[0] + p[1]) + (p[2] + p[3])) +
                      ((p[4] + p[5]) + (p[6] + p[7])) + ob;
            yreg += cut * tanhf(o * dt_over_cut);
            ypad[col] = (__bf16)yreg;   // col-based: never touches pad zeros
            if (wave == 0)
                out[((size_t)b * C_DIM + col) * (size_t)T + t] = yreg;
        }
    }
}

extern "C" void kernel_launch(void* const* d_in, const int* in_sizes, int n_in,
                              void* d_out, int out_size, void* d_ws, size_t ws_size,
                              hipStream_t stream) {
    const float* y0   = (const float*)d_in[0];
    const float* inW  = (const float*)d_in[1];
    const float* inB  = (const float*)d_in[2];
    const float* outW = (const float*)d_in[3];
    const float* outB = (const float*)d_in[4];
    const float* pW   = (const float*)d_in[5];
    const float* pB   = (const float*)d_in[6];
    const float* cut  = (const float*)d_in[7];
    const int*   Tp   = (const int*)d_in[8];

    const int B = in_sizes[0] / C_DIM;   // y0 is [B, C]

    decoder_rollout_kernel<<<dim3(B), dim3(256), 0, stream>>>(
        y0, inW, inB, outW, outB, pW, pB, cut, Tp, (float*)d_out);
}
